// StepWiseVAEMLPAutoEncoder_46737834115441
// MI455X (gfx1250) — matrix operand order repaired and checked by reading, not compiled, not run on hardware
//
#include <hip/hip_runtime.h>
#include <math.h>

// ---------------------------------------------------------------------------
// Types for CDNA5 WMMA (wave32, V_WMMA_F32_16X16X32_BF16)
// ---------------------------------------------------------------------------
typedef __bf16 bf16;
typedef __bf16 v8bf  __attribute__((ext_vector_type(8)));
typedef __bf16 v16bf __attribute__((ext_vector_type(16)));
typedef float  v8f   __attribute__((ext_vector_type(8)));

union ABFrag { v16bf v; v8bf h[2]; };

#define GEMM_WAVES 8

// ---------------------------------------------------------------------------
// GEMM argument block (passed by value)
// ---------------------------------------------------------------------------
struct GemmArgs {
  const bf16* A;    int lda;    // plain bf16 A [M, Kp]
  const bf16* Wt;   int ldw;    // transposed padded weights [Np, Kp]
  const float* bias;            // padded bias [Np]
  bf16* outB;       int ldoB;   // bf16 output
  float* outF;      long long ldoFr; int ldoFc; // f32 output (strided)
  int M, Np, Kp;
  // composite-A sources
  const float* x;               // encoder: raw input [B, 1024, 256] f32
  const bf16* ctrl;             // decoder: vdec output [64, 8192] bf16
  const bf16* prev;             // decoder: prev stft [64, 1024] bf16
  int t; float tval;
};

__device__ inline v8bf zero_v8bf() {
  v8bf z;
#pragma unroll
  for (int i = 0; i < 8; ++i) z[i] = (bf16)0.0f;
  return z;
}

// Load 8 contiguous (in K) bf16 A elements for matrix row `row`, col base `kc`
// (kc is always a multiple of 8; all section boundaries are multiples of 8).
template<int AL>
__device__ inline v8bf loadA8(const GemmArgs& g, int row, int kc) {
  if (AL == 0) {                       // plain bf16 activation buffer
    return *(const v8bf*)(g.A + (long long)row * g.lda + kc);
  } else if (AL == 1) {                // decoder step: [ctrl(32)|prev(1024)|t|0...]
    if (kc < 32)   return *(const v8bf*)(g.ctrl + row * 8192 + g.t * 32 + kc);
    if (kc < 1056) return *(const v8bf*)(g.prev + row * 1024 + (kc - 32));
    v8bf r = zero_v8bf();
    if (kc == 1056) r[0] = (bf16)g.tval;
    return r;
  } else {                             // encoder input: row=b*256+t, [prev|curr|t|0...]
    const int b = row >> 8, t = row & 255;
    v8bf r;
    if (kc < 1024) {                   // prev_stft = x[b, kc.., t-1] (zeros at t==0)
      if (t == 0) return zero_v8bf();
      const float* p = g.x + (long long)b * 262144 + (long long)kc * 256 + (t - 1);
#pragma unroll
      for (int i = 0; i < 8; ++i) r[i] = (bf16)p[i * 256];
      return r;
    } else if (kc < 2048) {            // curr_stft = x[b, kc-1024.., t]
      const float* p = g.x + (long long)b * 262144 + (long long)(kc - 1024) * 256 + t;
#pragma unroll
      for (int i = 0; i < 8; ++i) r[i] = (bf16)p[i * 256];
      return r;
    } else {
      r = zero_v8bf();
      if (kc == 2048) r[0] = (bf16)((float)t * (1.0f / 256.0f));
      return r;
    }
  }
}

// ---------------------------------------------------------------------------
// Universal bf16 WMMA GEMM: out = act(A[M,Kp] * Wt[Np,Kp]^T + bias)
// One wave computes a 16(M) x 32(N) tile = 2 accumulators, A frag reused.
// OM: 0 = bf16 out + ReLU, 1 = bf16 out, 2 = f32 strided out,
//     3 = f32 strided out + bf16 out (decoder layer 3: rec column + prev_stft)
// ---------------------------------------------------------------------------
template<int AL, int OM>
__global__ void __launch_bounds__(32 * GEMM_WAVES)
gemm_k(GemmArgs g) {
  const int lane = threadIdx.x & 31;
  const int wave = threadIdx.x >> 5;
  const int nT32 = g.Np >> 5;
  const int total = (g.M >> 4) * nT32;
  const int tile = blockIdx.x * GEMM_WAVES + wave;
  if (tile >= total) return;          // whole-wave uniform exit (EXEC all-1 for WMMA)
  const int mT = tile / nT32;
  const int nT = tile % nT32;
  const int lh = lane & 15;
  const int half = lane >> 4;
  const int row = mT * 16 + lh;       // A row for this lane (A-frag: lane%16 = M)
  const bf16* w0 = g.Wt + (long long)(nT * 32 + lh) * g.ldw;       // B-frag: lane%16 = N
  const bf16* w1 = g.Wt + (long long)(nT * 32 + 16 + lh) * g.ldw;

  v8f acc0, acc1;
#pragma unroll
  for (int i = 0; i < 8; ++i) { acc0[i] = 0.0f; acc1[i] = 0.0f; }

  for (int k = 0; k < g.Kp; k += 32) {
    ABFrag a, b0, b1;
    // A 16x32 bf16 layout: lanes0-15 K=0..7,16..23 ; lanes16-31 K=8..15,24..31
    a.h[0] = loadA8<AL>(g, row, k + half * 8);
    a.h[1] = loadA8<AL>(g, row, k + 16 + half * 8);
    // B 32x16 bf16 layout: lanes0-15 K=0..7,16..23 ; lanes16-31 K=8..15,24..31 (same as A)
    const int kb = k + half * 8;
    b0.h[0] = *(const v8bf*)(w0 + kb);
    b0.h[1] = *(const v8bf*)(w0 + kb + 16);
    b1.h[0] = *(const v8bf*)(w1 + kb);
    b1.h[1] = *(const v8bf*)(w1 + kb + 16);
    acc0 = __builtin_amdgcn_wmma_f32_16x16x32_bf16(false, a.v, false, b0.v,
                                                   (short)0, acc0, false, false);
    acc1 = __builtin_amdgcn_wmma_f32_16x16x32_bf16(false, a.v, false, b1.v,
                                                   (short)0, acc1, false, false);
  }

  // C/D layout: lane holds column n = lane%16, rows r + 8*(lane/16)
  const int nc0 = nT * 32 + lh;
  const int nc1 = nc0 + 16;
  const float bias0 = g.bias[nc0];
  const float bias1 = g.bias[nc1];
#pragma unroll
  for (int r = 0; r < 8; ++r) {
    const int m = mT * 16 + half * 8 + r;
    float v0 = acc0[r] + bias0;
    float v1 = acc1[r] + bias1;
    if (OM == 0) { v0 = fmaxf(v0, 0.0f); v1 = fmaxf(v1, 0.0f); }
    if (OM == 0 || OM == 1 || OM == 3) {
      g.outB[(long long)m * g.ldoB + nc0] = (bf16)v0;
      g.outB[(long long)m * g.ldoB + nc1] = (bf16)v1;
    }
    if (OM == 2 || OM == 3) {
      g.outF[(long long)m * g.ldoFr + (long long)nc0 * g.ldoFc] = v0;
      g.outF[(long long)m * g.ldoFr + (long long)nc1 * g.ldoFc] = v1;
    }
  }
}

// ---------------------------------------------------------------------------
// Prep kernels
// ---------------------------------------------------------------------------
__global__ void transpose_pad_k(const float* __restrict__ W, bf16* __restrict__ Wt,
                                int K, int N, int Kp, int Np) {
  const long long idx = (long long)blockIdx.x * blockDim.x + threadIdx.x;
  const long long tot = (long long)Kp * Np;
  if (idx >= tot) return;
  const int k = (int)(idx % Kp);
  const int n = (int)(idx / Kp);
  const float v = (k < K && n < N) ? W[(long long)k * N + n] : 0.0f;
  Wt[idx] = (bf16)v;
}

__global__ void pad_bias_k(const float* __restrict__ b, float* __restrict__ bp,
                           int N, int Np) {
  const int i = blockIdx.x * blockDim.x + threadIdx.x;
  if (i < Np) bp[i] = (i < N) ? b[i] : 0.0f;
}

__global__ void zero_bf16_k(bf16* p, int n) {
  const int i = blockIdx.x * blockDim.x + threadIdx.x;
  if (i < n) p[i] = (bf16)0.0f;
}

__global__ void reparam_k(const float* __restrict__ mu, const float* __restrict__ lv,
                          const float* __restrict__ eps, bf16* __restrict__ z, int n) {
  const int i = blockIdx.x * blockDim.x + threadIdx.x;
  if (i < n) z[i] = (bf16)(mu[i] + eps[i] * expf(0.5f * lv[i]));
}

// ---------------------------------------------------------------------------
// Host orchestration
// ---------------------------------------------------------------------------
extern "C" void kernel_launch(void* const* d_in, const int* in_sizes, int n_in,
                              void* d_out, int out_size, void* d_ws, size_t ws_size,
                              hipStream_t stream) {
  (void)in_sizes; (void)n_in; (void)out_size; (void)ws_size;
  const float* x   = (const float*)d_in[0];
  const float* eps = (const float*)d_in[1];

  auto pad32 = [](int n) { return (n + 31) & ~31; };

  size_t off = 0;
  auto alloc = [&](size_t bytes) -> void* {
    off = (off + 255) & ~(size_t)255;
    void* p = (char*)d_ws + off;
    off += bytes;
    return p;
  };

  // 13 linear layers, flattened pytree order: x, eps, enc(3), dec(3), venc(2),
  // mu, logvar, vdec(3) with (W, b) leaves.
  struct L { int wi, K, N; };
  const L layers[13] = {
    {2, 2049, 512}, {4, 512, 128}, {6, 128, 32},          // encoder MLP
    {8, 1057, 1046}, {10, 1046, 1035}, {12, 1035, 1024},  // decoder MLP
    {14, 8192, 1625}, {16, 1625, 323},                    // VAE encoder
    {18, 323, 64}, {20, 323, 64},                         // mu, logvar
    {22, 64, 323}, {24, 323, 1625}, {26, 1625, 8192},     // VAE decoder
  };
  bf16* wt[13]; float* bp[13];
  for (int i = 0; i < 13; ++i) {
    const int Kp = pad32(layers[i].K), Np = pad32(layers[i].N);
    wt[i] = (bf16*)alloc((size_t)Kp * Np * sizeof(bf16));
    bp[i] = (float*)alloc((size_t)Np * sizeof(float));
  }
  bf16* encH1  = (bf16*)alloc((size_t)16384 * 512 * sizeof(bf16));
  bf16* encH2  = (bf16*)alloc((size_t)16384 * 128 * sizeof(bf16));
  bf16* ctrl   = (bf16*)alloc((size_t)16384 * 32 * sizeof(bf16)); // == [64, 8192]
  bf16* vencH1 = (bf16*)alloc((size_t)64 * 1632 * sizeof(bf16));
  bf16* vencH2 = (bf16*)alloc((size_t)64 * 352 * sizeof(bf16));
  bf16* zbuf   = (bf16*)alloc((size_t)64 * 64 * sizeof(bf16));
  bf16* vdH1   = (bf16*)alloc((size_t)64 * 352 * sizeof(bf16));
  bf16* vdH2   = (bf16*)alloc((size_t)64 * 1632 * sizeof(bf16));
  bf16* vdOut  = (bf16*)alloc((size_t)64 * 8192 * sizeof(bf16));
  bf16* decH1  = (bf16*)alloc((size_t)64 * 1056 * sizeof(bf16));
  bf16* decH2  = (bf16*)alloc((size_t)64 * 1056 * sizeof(bf16));
  bf16* prev   = (bf16*)alloc((size_t)64 * 1024 * sizeof(bf16));

  // ---- weight prep: f32 [K,N] -> bf16 transposed padded [Np, Kp]
  for (int i = 0; i < 13; ++i) {
    const int K = layers[i].K, N = layers[i].N;
    const int Kp = pad32(K), Np = pad32(N);
    const long long tot = (long long)Kp * Np;
    transpose_pad_k<<<(int)((tot + 255) / 256), 256, 0, stream>>>(
        (const float*)d_in[layers[i].wi], wt[i], K, N, Kp, Np);
    pad_bias_k<<<(Np + 255) / 256, 256, 0, stream>>>(
        (const float*)d_in[layers[i].wi + 1], bp[i], N, Np);
  }
  zero_bf16_k<<<(64 * 1024 + 255) / 256, 256, 0, stream>>>(prev, 64 * 1024);

  float* rec   = (float*)d_out;               // [64, 1024, 256]
  float* muOut = rec + 16777216;              // [64, 64]
  float* lvOut = muOut + 4096;                // [64, 64]

  auto run = [&](int AL, int OM, const GemmArgs& g) {
    const int tiles = (g.M >> 4) * (g.Np >> 5);
    const int blocks = (tiles + GEMM_WAVES - 1) / GEMM_WAVES;
    dim3 gr(blocks), bl(32 * GEMM_WAVES);
    switch (AL * 10 + OM) {
      case 0:  gemm_k<0, 0><<<gr, bl, 0, stream>>>(g); break;
      case 1:  gemm_k<0, 1><<<gr, bl, 0, stream>>>(g); break;
      case 2:  gemm_k<0, 2><<<gr, bl, 0, stream>>>(g); break;
      case 3:  gemm_k<0, 3><<<gr, bl, 0, stream>>>(g); break;
      case 10: gemm_k<1, 0><<<gr, bl, 0, stream>>>(g); break;
      case 20: gemm_k<2, 0><<<gr, bl, 0, stream>>>(g); break;
    }
  };

  // ---- Phase 1: batched encoder MLP over B*T = 16384 rows (A fused from x)
  { GemmArgs g{}; g.x = x; g.Wt = wt[0]; g.ldw = 2080; g.bias = bp[0];
    g.outB = encH1; g.ldoB = 512; g.M = 16384; g.Np = 512; g.Kp = 2080; run(2, 0, g); }
  { GemmArgs g{}; g.A = encH1; g.lda = 512; g.Wt = wt[1]; g.ldw = 512; g.bias = bp[1];
    g.outB = encH2; g.ldoB = 128; g.M = 16384; g.Np = 128; g.Kp = 512; run(0, 0, g); }
  { GemmArgs g{}; g.A = encH2; g.lda = 128; g.Wt = wt[2]; g.ldw = 128; g.bias = bp[2];
    g.outB = ctrl; g.ldoB = 32; g.M = 16384; g.Np = 32; g.Kp = 128; run(0, 1, g); }

  // ---- Phase 2: VAE encode (ctrl buffer [16384,32] == [64,8192] row-major)
  { GemmArgs g{}; g.A = ctrl; g.lda = 8192; g.Wt = wt[6]; g.ldw = 8192; g.bias = bp[6];
    g.outB = vencH1; g.ldoB = 1632; g.M = 64; g.Np = 1632; g.Kp = 8192; run(0, 0, g); }
  { GemmArgs g{}; g.A = vencH1; g.lda = 1632; g.Wt = wt[7]; g.ldw = 1632; g.bias = bp[7];
    g.outB = vencH2; g.ldoB = 352; g.M = 64; g.Np = 352; g.Kp = 1632; run(0, 0, g); }
  { GemmArgs g{}; g.A = vencH2; g.lda = 352; g.Wt = wt[8]; g.ldw = 352; g.bias = bp[8];
    g.outF = muOut; g.ldoFr = 64; g.ldoFc = 1; g.M = 64; g.Np = 64; g.Kp = 352; run(0, 2, g); }
  { GemmArgs g{}; g.A = vencH2; g.lda = 352; g.Wt = wt[9]; g.ldw = 352; g.bias = bp[9];
    g.outF = lvOut; g.ldoFr = 64; g.ldoFc = 1; g.M = 64; g.Np = 64; g.Kp = 352; run(0, 2, g); }

  // ---- Phase 3: reparameterize z = mu + eps*exp(0.5*logvar)
  reparam_k<<<16, 256, 0, stream>>>(muOut, lvOut, eps, zbuf, 4096);

  // ---- Phase 4: VAE decode
  { GemmArgs g{}; g.A = zbuf; g.lda = 64; g.Wt = wt[10]; g.ldw = 64; g.bias = bp[10];
    g.outB = vdH1; g.ldoB = 352; g.M = 64; g.Np = 352; g.Kp = 64; run(0, 0, g); }
  { GemmArgs g{}; g.A = vdH1; g.lda = 352; g.Wt = wt[11]; g.ldw = 352; g.bias = bp[11];
    g.outB = vdH2; g.ldoB = 1632; g.M = 64; g.Np = 1632; g.Kp = 352; run(0, 0, g); }
  { GemmArgs g{}; g.A = vdH2; g.lda = 1632; g.Wt = wt[12]; g.ldw = 1632; g.bias = bp[12];
    g.outB = vdOut; g.ldoB = 8192; g.M = 64; g.Np = 8192; g.Kp = 1632; run(0, 1, g); }

  // ---- Phase 5: sequential decoder scan (256 steps x 3 layers)
  for (int t = 0; t < 256; ++t) {
    { GemmArgs g{}; g.ctrl = vdOut; g.prev = prev; g.t = t;
      g.tval = (float)t * (1.0f / 256.0f);
      g.Wt = wt[3]; g.ldw = 1088; g.bias = bp[3];
      g.outB = decH1; g.ldoB = 1056; g.M = 64; g.Np = 1056; g.Kp = 1088; run(1, 0, g); }
    { GemmArgs g{}; g.A = decH1; g.lda = 1056; g.Wt = wt[4]; g.ldw = 1056; g.bias = bp[4];
      g.outB = decH2; g.ldoB = 1056; g.M = 64; g.Np = 1056; g.Kp = 1056; run(0, 0, g); }
    { GemmArgs g{}; g.A = decH2; g.lda = 1056; g.Wt = wt[5]; g.ldw = 1056; g.bias = bp[5];
      g.outB = prev; g.ldoB = 1024;                    // feedback for step t+1
      g.outF = rec + t; g.ldoFr = 262144; g.ldoFc = 256; // rec[b, s, t]
      g.M = 64; g.Np = 1024; g.Kp = 1056; run(0, 3, g); }
  }
}
